// WireframeDetector_67568425500716
// MI455X (gfx1250) — matrix operand order repaired and checked by reading, not compiled, not run on hardware
//
#include <hip/hip_runtime.h>
#include <math.h>

#define HGT 256
#define WID 256
#define NPIX 65536
#define NJ 300
#define NL 4096
#define NLP 196608          // 3*H*W proposal lines
#define DLOI 128
#define DFC 1024
#define SENT 90000          // NJ*NJ
#define NWORDS 2813         // ceil(90000/32)
#define PI_F 3.14159265358979f

typedef __attribute__((ext_vector_type(16))) _Float16 v16h;
typedef __attribute__((ext_vector_type(8)))  float    v8f;

__device__ __forceinline__ float sigmoidf(float x){ return 1.0f/(1.0f+expf(-x)); }

// wave-relative LDS byte offset from a generic pointer to a __shared__ object:
// flat LDS addresses carry the aperture in the high 32 bits, offset in the low 32.
__device__ __forceinline__ unsigned lds_off(const void* p){
    return (unsigned)(unsigned long long)p;
}

// ---------------- init: zero bitmap + used ----------------
__global__ void k_init(unsigned* bitmap, int* used_i){
    int t = blockIdx.x*blockDim.x + threadIdx.x;
    if (t < NWORDS+3) bitmap[t] = 0u;
    if (t < NJ) used_i[t] = 0;
}

// ---------------- f32 -> f16 conversion ----------------
__global__ void k_cvt(const float* __restrict__ s, _Float16* __restrict__ d, int n){
    for (int i = blockIdx.x*blockDim.x + threadIdx.x; i < n; i += gridDim.x*blockDim.x)
        d[i] = (_Float16)s[i];
}

// ---------------- preprocess: jloc + proposal lines ----------------
__global__ void k_pre(const float* __restrict__ o, float* __restrict__ jloc,
                      float4* __restrict__ lines_pred){
    int p = blockIdx.x*blockDim.x + threadIdx.x;
    if (p >= NPIX) return;
    float o0=o[p], o1=o[NPIX+p], o2=o[2*NPIX+p], o3=o[3*NPIX+p], o4=o[4*NPIX+p];
    float o5=o[5*NPIX+p], o6=o[6*NPIX+p];
    jloc[p] = sigmoidf(o6 - o5);              // softmax ch1 of 2 == sigmoid(diff)
    float md = (sigmoidf(o0) - 0.5f) * 2.0f * PI_F;
    float st = sigmoidf(o1) * PI_F * 0.5f;
    float ed = -sigmoidf(o2) * PI_F * 0.5f;
    float cs_md = cosf(md), ss_md = sinf(md);
    float cs_st = fmaxf(cosf(st), 0.001f), ss_st = fmaxf(sinf(st), 0.001f);
    float cs_ed = fmaxf(cosf(ed), 0.001f), ss_ed = fminf(sinf(ed), -0.001f);
    float y_st = ss_st/cs_st, y_ed = ss_ed/cs_ed;
    float dis = sigmoidf(o3), res = sigmoidf(o4);
    float x0 = (float)(p % WID), y0 = (float)(p / WID);
    float ax = cs_md - ss_md*y_st, ay = ss_md + cs_md*y_st;
    float bx = cs_md - ss_md*y_ed, by = ss_md + cs_md*y_ed;
    #pragma unroll
    for (int s = 0; s < 3; s++){
        float dn = (dis + (float)(s-1)*res) * 5.0f;
        float xs = fminf(fmaxf(ax*dn + x0, 0.f), 255.f);
        float ys = fminf(fmaxf(ay*dn + y0, 0.f), 255.f);
        float xe = fminf(fmaxf(bx*dn + x0, 0.f), 255.f);
        float ye = fminf(fmaxf(by*dn + y0, 0.f), 255.f);
        lines_pred[s*NPIX + p] = make_float4(xs, ys, xe, ye);
    }
}

// ---------------- 3x3 NMS ----------------
__global__ void k_nms(const float* __restrict__ jloc, float* __restrict__ jnms){
    int p = blockIdx.x*blockDim.x + threadIdx.x;
    if (p >= NPIX) return;
    int x = p % WID, y = p / WID;
    float a = jloc[p], m = a;
    for (int dy = -1; dy <= 1; dy++){
        int yy = y + dy; if (yy < 0 || yy >= HGT) continue;
        for (int dx = -1; dx <= 1; dx++){
            int xx = x + dx; if (xx < 0 || xx >= WID) continue;
            m = fmaxf(m, jloc[yy*WID + xx]);
        }
    }
    jnms[p] = (a == m) ? a : 0.0f;
}

// ---------------- exact top-300 (single block, packed-key reduce) ----------------
__global__ void k_topk(const float* __restrict__ o, float* jnms, int* jidx, float* jsc,
                       float* jx, float* jy, float* jpad,
                       float* out_jsc, float* out_juncs){
    __shared__ unsigned long long red[1024];
    int tid = threadIdx.x;
    for (int t = 0; t < NJ; t++){
        unsigned long long best = 0ull;
        for (int i = tid; i < NPIX; i += 1024){
            float f = jnms[i];
            unsigned u = __float_as_uint(f);
            u = (u & 0x80000000u) ? ~u : (u | 0x80000000u);
            unsigned long long key = ((unsigned long long)u << 32) | (unsigned)(~i);
            if (key > best) best = key;
        }
        red[tid] = best; __syncthreads();
        for (int off = 512; off > 0; off >>= 1){
            if (tid < off) red[tid] = red[tid] > red[tid+off] ? red[tid] : red[tid+off];
            __syncthreads();
        }
        if (tid == 0){
            int idx = (int)(~(unsigned)(red[0] & 0xffffffffull));
            jidx[t] = idx; jsc[t] = jnms[idx];
            jnms[idx] = -1e30f;
        }
        __syncthreads();
    }
    if (tid < NJ){
        int idx = jidx[tid]; float score = jsc[tid];
        float ox = sigmoidf(o[7*NPIX + idx]) - 0.5f;
        float oy = sigmoidf(o[8*NPIX + idx]) - 0.5f;
        float fx = (float)(idx % WID) + ox + 0.5f;
        float fy = (float)(idx / WID) + oy + 0.5f;
        jx[tid] = fx; jy[tid] = fy;
        jpad[tid] = (score > 0.008f) ? 0.0f : 1e12f;
        out_juncs[tid*2+0] = fx * (1.0f/WID);
        out_juncs[tid*2+1] = fy * (1.0f/HGT);
        out_jsc[tid] = score;
    }
}

// ---------------- line<->junction matching + bitmap of unique pair keys ----------------
__global__ void k_match(const float4* __restrict__ lines_pred, const float* __restrict__ jx,
                        const float* __restrict__ jy, const float* __restrict__ jpad,
                        unsigned* bitmap){
    __shared__ float sx[NJ], sy[NJ], sp[NJ];
    for (int j = threadIdx.x; j < NJ; j += blockDim.x){ sx[j]=jx[j]; sy[j]=jy[j]; sp[j]=jpad[j]; }
    __syncthreads();
    int l = blockIdx.x*blockDim.x + threadIdx.x;
    if (l >= NLP) return;
    float4 L = lines_pred[l];
    float b1 = 3.4e38f, b2 = 3.4e38f; int i1 = 0, i2 = 0;
    for (int j = 0; j < NJ; j++){
        float dx1 = L.x - sx[j], dy1 = L.y - sy[j];
        float d1 = dx1*dx1 + dy1*dy1 + sp[j];
        float dx2 = L.z - sx[j], dy2 = L.w - sy[j];
        float d2 = dx2*dx2 + dy2*dy2 + sp[j];
        if (d1 < b1){ b1 = d1; i1 = j; }
        if (d2 < b2){ b2 = d2; i2 = j; }
    }
    int mn = min(i1, i2), mx = max(i1, i2);
    if (mn < mx){
        int key = mn*NJ + mx;
        atomicOr(&bitmap[key >> 5], 1u << (key & 31));
    }
}

// ---------------- bitmap prefix-scan -> sorted unique first-4096 pairs ----------------
__global__ void k_pairs(const unsigned* __restrict__ bitmap, int* pair, float* out_np){
    __shared__ int pre[1024];
    __shared__ int s_clamped;
    int tid = threadIdx.x;
    int w0 = tid*3, cnt = 0;
    for (int i = 0; i < 3; i++){ int w = w0+i; if (w < NWORDS) cnt += __popc(bitmap[w]); }
    pre[tid] = cnt; __syncthreads();
    for (int off = 1; off < 1024; off <<= 1){
        int v = (tid >= off) ? pre[tid-off] : 0; __syncthreads();
        pre[tid] += v; __syncthreads();
    }
    int total = pre[1023];
    int rank = pre[tid] - cnt;          // exclusive prefix
    for (int i = 0; i < 3; i++){
        int w = w0+i; if (w >= NWORDS) break;
        unsigned bits = bitmap[w];
        while (bits){
            int b = __ffs(bits) - 1; bits &= bits - 1;
            if (rank < NL) pair[rank] = w*32 + b;
            rank++;
        }
    }
    if (tid == 0){ s_clamped = min(total, NL); out_np[0] = (float)min(total, NL); }
    __syncthreads();
    for (int r = tid; r < NL; r += 1024) if (r >= s_clamped) pair[r] = SENT;
}

// ---------------- adjusted lines + scaled output + used flags ----------------
__global__ void k_adj(const int* __restrict__ pair, const float* __restrict__ jx,
                      const float* __restrict__ jy, float4* __restrict__ lines_adj,
                      float* __restrict__ out_lines, int* used_i){
    int l = blockIdx.x*blockDim.x + threadIdx.x;
    if (l >= NL) return;
    int k = pair[l]; bool valid = (k < SENT);
    int p1 = valid ? k / NJ : 0, p2 = valid ? k % NJ : 0;
    float x1 = jx[p1], y1 = jy[p1], x2 = jx[p2], y2 = jy[p2];
    lines_adj[l] = make_float4(x1, y1, x2, y2);
    out_lines[l*4+0] = x1 * (1.0f/WID);
    out_lines[l*4+1] = y1 * (1.0f/HGT);
    out_lines[l*4+2] = x2 * (1.0f/WID);
    out_lines[l*4+3] = y2 * (1.0f/HGT);
    if (valid){ atomicMax(&used_i[p1], 1); atomicMax(&used_i[p2], 1); }
}

// ---- async-stage a 32(K) x 16(N) f16 B tile into LDS (K-major, 64x16B chunks) ----
// Each lane issues two GLOBAL_LOAD_ASYNC_TO_LDS_B128 (coalesced 16B row chunks),
// tracked by ASYNCcnt. Caller brackets with s_wait_dscnt / s_wait_asynccnt.
__device__ __forceinline__ void stage_B(const _Float16* B, int ldb, int kk, int n0,
                                        unsigned lbase, int lane){
    #pragma unroll
    for (int t = 0; t < 2; t++){
        int c = lane + t*32;                 // chunk id 0..63
        int k = c >> 1, piece = c & 1;       // row, 8-half piece
        unsigned long long g =
            (unsigned long long)(B + (size_t)(kk + k)*ldb + n0 + piece*8);
        unsigned l = lbase + (unsigned)c*16u;
        asm volatile("global_load_async_to_lds_b128 %0, %1, off"
                     :: "v"(l), "v"(g) : "memory");
    }
}

// ---------------- loi GEMM: (128x256) @ (256x65536) + bias, WMMA f16 ----------------
// One wave per 16-wide N tile; all 8 M-tiles reuse the LDS-staged B fragment.
__global__ void k_loi(const _Float16* __restrict__ Wf, const _Float16* __restrict__ F,
                      const float* __restrict__ bias, float* __restrict__ loi){
    __shared__ _Float16 sB[32*16];
    int lane = threadIdx.x;
    int nloc = lane & 15, hi = lane >> 4;
    int n0 = blockIdx.x * 16;
    unsigned lbase = lds_off(sB);
    v8f zero = {0.f,0.f,0.f,0.f,0.f,0.f,0.f,0.f};
    v8f acc[8];
    #pragma unroll
    for (int mt = 0; mt < 8; mt++) acc[mt] = zero;
    for (int kk = 0; kk < 256; kk += 32){
        asm volatile("s_wait_dscnt 0x0" ::: "memory");      // prev frag reads done
        stage_B(F, NPIX, kk, n0, lbase, lane);
        asm volatile("s_wait_asynccnt 0x0" ::: "memory");   // tile landed in LDS
        v16h b;
        #pragma unroll
        for (int h = 0; h < 16; h++) b[h] = sB[(hi*8 + (h&7) + (h>>3)*16)*16 + nloc];
        #pragma unroll
        for (int mt = 0; mt < 8; mt++){
            int m = mt*16 + nloc;
            const _Float16* Ar = Wf + m*256 + kk + hi*8;
            v16h a;
            #pragma unroll
            for (int h = 0; h < 8; h++){ a[h] = Ar[h]; a[h+8] = Ar[16 + h]; }
            acc[mt] = __builtin_amdgcn_wmma_f32_16x16x32_f16(
                false, a, false, b, (short)0, acc[mt], false, false);
        }
    }
    #pragma unroll
    for (int mt = 0; mt < 8; mt++)
        #pragma unroll
        for (int r = 0; r < 8; r++){
            int m = mt*16 + r + hi*8;
            loi[m*NPIX + n0 + nloc] = acc[mt][r] + bias[m];
        }
}

// ---------------- bilinear sampling + maxpool -> x (f16) ----------------
__global__ void k_pool(const float4* __restrict__ lines_adj, const float* __restrict__ loi,
                       _Float16* __restrict__ xh){
    __shared__ int s00[32], s10[32], s01[32], s11[32];
    __shared__ float w00[32], w10[32], w01[32], w11[32];
    int l = blockIdx.x;
    if (threadIdx.x < 32){
        int i = threadIdx.x;
        float4 L = lines_adj[l];
        float t = (float)i / 31.0f;
        float px = L.x*t + L.z*(1.0f-t) - 0.5f;
        float py = L.y*t + L.w*(1.0f-t) - 0.5f;
        float px0 = fminf(fmaxf(floorf(px), 0.f), 255.f);
        float py0 = fminf(fmaxf(floorf(py), 0.f), 255.f);
        float px1 = fminf(px0 + 1.0f, 255.f);
        float py1 = fminf(py0 + 1.0f, 255.f);
        int ix0=(int)px0, iy0=(int)py0, ix1=(int)px1, iy1=(int)py1;
        s00[i]=iy0*WID+ix0; s10[i]=iy1*WID+ix0; s01[i]=iy0*WID+ix1; s11[i]=iy1*WID+ix1;
        w00[i]=(py1-py)*(px1-px); w10[i]=(py-py0)*(px1-px);
        w01[i]=(py1-py)*(px-px0); w11[i]=(py-py0)*(px-px0);
    }
    __syncthreads();
    int c = threadIdx.x;           // 128 channels
    const float* lc = loi + (size_t)c * NPIX;
    #pragma unroll
    for (int j = 0; j < 8; j++){
        float m = -3.4e38f;
        #pragma unroll
        for (int q = 0; q < 4; q++){
            int i = j*4 + q;
            float v = lc[s00[i]]*w00[i] + lc[s10[i]]*w10[i]
                    + lc[s01[i]]*w01[i] + lc[s11[i]]*w11[i];
            m = fmaxf(m, v);
        }
        xh[(size_t)l*DFC + c*8 + j] = (_Float16)m;
    }
}

// ---------------- generic WMMA f16 FC: D = act(A@B + bias) ----------------
// one wave per (128-row strip x 16-col tile); async-staged B fragment reused by 8 M-tiles
__global__ void k_fc(const _Float16* __restrict__ A, const _Float16* __restrict__ B,
                     const float* __restrict__ bias, _Float16* __restrict__ D,
                     int M, int N, int K, int relu){
    __shared__ _Float16 sB[32*16];
    int lane = threadIdx.x;
    int nloc = lane & 15, hi = lane >> 4;
    int n0 = blockIdx.x * 16;
    int m0 = blockIdx.y * 128;
    unsigned lbase = lds_off(sB);
    v8f zero = {0.f,0.f,0.f,0.f,0.f,0.f,0.f,0.f};
    v8f acc[8];
    #pragma unroll
    for (int mt = 0; mt < 8; mt++) acc[mt] = zero;
    for (int kk = 0; kk < K; kk += 32){
        asm volatile("s_wait_dscnt 0x0" ::: "memory");
        stage_B(B, N, kk, n0, lbase, lane);
        asm volatile("s_wait_asynccnt 0x0" ::: "memory");
        v16h b;
        #pragma unroll
        for (int h = 0; h < 16; h++) b[h] = sB[(hi*8 + (h&7) + (h>>3)*16)*16 + nloc];
        #pragma unroll
        for (int mt = 0; mt < 8; mt++){
            int m = m0 + mt*16 + nloc;
            const _Float16* Ar = A + (size_t)m*K + kk + hi*8;
            v16h a;
            #pragma unroll
            for (int h = 0; h < 8; h++){ a[h] = Ar[h]; a[h+8] = Ar[16 + h]; }
            acc[mt] = __builtin_amdgcn_wmma_f32_16x16x32_f16(
                false, a, false, b, (short)0, acc[mt], false, false);
        }
    }
    float bn = bias[n0 + nloc];
    #pragma unroll
    for (int mt = 0; mt < 8; mt++)
        #pragma unroll
        for (int r = 0; r < 8; r++){
            int m = m0 + mt*16 + r + hi*8;
            float v = acc[mt][r] + bn;
            if (relu) v = fmaxf(v, 0.0f);
            D[(size_t)m*N + n0 + nloc] = (_Float16)v;
        }
}

// ---------------- final layer + sigmoid + mask ----------------
__global__ void k_score(const _Float16* __restrict__ h2, const float* __restrict__ w3,
                        const float* __restrict__ b3, const int* __restrict__ pair,
                        float* out_scores, float* out_mask){
    int l = blockIdx.x*blockDim.x + threadIdx.x;
    if (l >= NL) return;
    float s = b3[0];
    const _Float16* r = h2 + (size_t)l*DFC;
    for (int k = 0; k < DFC; k++) s += (float)r[k] * w3[k];
    float sc = sigmoidf(s);
    out_scores[l] = sc;
    out_mask[l] = (pair[l] < SENT && sc > 0.05f) ? 1.0f : 0.0f;
}

__global__ void k_used(const int* used_i, float* out_used){
    int t = blockIdx.x*blockDim.x + threadIdx.x;
    if (t < NJ) out_used[t] = (used_i[t] > 0) ? 1.0f : 0.0f;
}

// ---------------- host launch ----------------
static inline char* carve(char*& p, size_t bytes){
    char* r = p;
    p += (bytes + 255) & ~(size_t)255;
    return r;
}

extern "C" void kernel_launch(void* const* d_in, const int* in_sizes, int n_in,
                              void* d_out, int out_size, void* d_ws, size_t ws_size,
                              hipStream_t stream) {
    const float* output   = (const float*)d_in[0];
    const float* features = (const float*)d_in[1];
    const float* fc1_w    = (const float*)d_in[2];
    const float* fc1_b    = (const float*)d_in[3];
    const float* w1       = (const float*)d_in[4];
    const float* b1       = (const float*)d_in[5];
    const float* w2       = (const float*)d_in[6];
    const float* b2       = (const float*)d_in[7];
    const float* w3       = (const float*)d_in[8];
    const float* b3       = (const float*)d_in[9];

    float* out = (float*)d_out;
    float* o_lines  = out;              // 4096*4
    float* o_scores = out + 16384;      // 4096
    float* o_mask   = out + 20480;      // 4096
    float* o_juncs  = out + 24576;      // 300*2
    float* o_jsc    = out + 25176;      // 300
    float* o_used   = out + 25476;      // 300
    float* o_np     = out + 25776;      // 1

    char* p = (char*)d_ws;
    float4*   lines_pred = (float4*)  carve(p, (size_t)NLP*16);
    float*    jloc       = (float*)   carve(p, (size_t)NPIX*4);
    float*    jnms       = (float*)   carve(p, (size_t)NPIX*4);
    int*      jidx       = (int*)     carve(p, NJ*4);
    float*    jsc        = (float*)   carve(p, NJ*4);
    float*    jx         = (float*)   carve(p, NJ*4);
    float*    jy         = (float*)   carve(p, NJ*4);
    float*    jpad       = (float*)   carve(p, NJ*4);
    unsigned* bitmap     = (unsigned*)carve(p, (NWORDS+3)*4);
    int*      pair       = (int*)     carve(p, NL*4);
    int*      used_i     = (int*)     carve(p, NJ*4);
    float4*   lines_adj  = (float4*)  carve(p, (size_t)NL*16);
    _Float16* feat_h     = (_Float16*)carve(p, (size_t)256*NPIX*2);
    _Float16* fw_h       = (_Float16*)carve(p, (size_t)DLOI*256*2);
    _Float16* w1_h       = (_Float16*)carve(p, (size_t)DFC*DFC*2);
    _Float16* w2_h       = (_Float16*)carve(p, (size_t)DFC*DFC*2);
    float*    loi        = (float*)   carve(p, (size_t)DLOI*NPIX*4);
    _Float16* x_h        = (_Float16*)carve(p, (size_t)NL*DFC*2);
    _Float16* h1_h       = (_Float16*)carve(p, (size_t)NL*DFC*2);
    _Float16* h2_h       = (_Float16*)carve(p, (size_t)NL*DFC*2);

    k_init<<<12, 256, 0, stream>>>(bitmap, used_i);

    k_cvt<<<4096, 256, 0, stream>>>(features, feat_h, 256*NPIX);
    k_cvt<<<128,  256, 0, stream>>>(fc1_w,   fw_h,   DLOI*256);
    k_cvt<<<4096, 256, 0, stream>>>(w1,      w1_h,   DFC*DFC);
    k_cvt<<<4096, 256, 0, stream>>>(w2,      w2_h,   DFC*DFC);

    k_pre<<<256, 256, 0, stream>>>(output, jloc, lines_pred);
    k_nms<<<256, 256, 0, stream>>>(jloc, jnms);
    k_topk<<<1, 1024, 0, stream>>>(output, jnms, jidx, jsc, jx, jy, jpad, o_jsc, o_juncs);
    k_match<<<768, 256, 0, stream>>>(lines_pred, jx, jy, jpad, bitmap);
    k_pairs<<<1, 1024, 0, stream>>>(bitmap, pair, o_np);
    k_adj<<<16, 256, 0, stream>>>(pair, jx, jy, lines_adj, o_lines, used_i);

    k_loi<<<NPIX/16, 32, 0, stream>>>(fw_h, feat_h, fc1_b, loi);
    k_pool<<<NL, 128, 0, stream>>>(lines_adj, loi, x_h);

    k_fc<<<dim3(DFC/16, NL/128), 32, 0, stream>>>(x_h,  w1_h, b1, h1_h, NL, DFC, DFC, 1);
    k_fc<<<dim3(DFC/16, NL/128), 32, 0, stream>>>(h1_h, w2_h, b2, h2_h, NL, DFC, DFC, 1);

    k_score<<<16, 256, 0, stream>>>(h2_h, w3, b3, pair, o_scores, o_mask);
    k_used<<<2, 256, 0, stream>>>(used_i, o_used);
}
